// FullAttention_25469156065949
// MI455X (gfx1250) — compile-verified, matrix operand order checked by reading, not run on hardware
//
#include <hip/hip_runtime.h>

// MI455X / gfx1250 wave32 WMMA attention forward (output + full attn matrix).
typedef __attribute__((ext_vector_type(16))) _Float16 v16h;
typedef __attribute__((ext_vector_type(8)))  float    v8f;

#define Bdim   8
#define Hdim   8
#define Sdim   1024
#define Ddim   64
#define BHdim  (Bdim * Hdim)
#define SCALE  0.125f            // 1/sqrt(64)

#define WAVES  8                 // 256 threads, wave32
#define QT     16                // q rows per wave
#define QBLK   (WAVES * QT)      // 128 q rows per block
#define KSTEP  32                // K/V rows staged per iteration
#define NKB    (Sdim / KSTEP)    // 32 K-tiles
#define LKF    68                // f32 K tile pitch (dwords): 68%64==4 -> conflict-free columns
#define LVPAD  40                // transposed-V row pitch (halves)
#define LPPAD  40                // P tile row pitch (halves)

// Per-lane element -> K/d index for 16-bit WMMA A (16x32) / B (32x16) fragments.
// v16h element e = VGPR (e/2), half (e&1).  ISA 7.12.2 layout.
// Contiguous for e=0..7 and e=8..15 -> gathers vectorize to ds_load_b128 pairs.
__device__ __forceinline__ int ab_idx(int lane, int e) {
  int v = e >> 1, h = e & 1;
  return ((v & 4) << 2) + ((v & 3) << 1) + h + ((lane >> 4) << 3);
}

// gfx1250 async global->LDS copy, 16B per lane, tracked by ASYNCcnt.
__device__ __forceinline__ void async_copy_b128(unsigned lds_off, const float* gaddr) {
  asm volatile("global_load_async_to_lds_b128 %0, %1, off"
               :: "v"(lds_off), "v"(gaddr) : "memory");
}
__device__ __forceinline__ void wait_async0() {           // all copies done
  asm volatile("s_wait_asynccnt 0x0" ::: "memory");
}
__device__ __forceinline__ void wait_async2() {           // current tile done, prefetch in flight
  asm volatile("s_wait_asynccnt 0x2" ::: "memory");
}

// Load the wave's 16xD Q tile as two A fragments (d 0..31, 32..63), scale folded in.
__device__ __forceinline__ void load_q_frags(const float* __restrict__ Qbh,
                                             int qbase, int lane, v16h& a0, v16h& a1) {
  const float* qr = Qbh + (size_t)(qbase + (lane & 15)) * Ddim;
#pragma unroll
  for (int e = 0; e < 16; ++e) {
    int d = ab_idx(lane, e);
    a0[e] = (_Float16)(qr[d] * SCALE);
    a1[e] = (_Float16)(qr[32 + d] * SCALE);
  }
}

// Score tile: C = (Q*scale) @ K^T for 16 q rows x 16 k cols (k-subtile t of f32 ldsK).
__device__ __forceinline__ v8f score_tile(const float (*ldsKf)[LKF],
                                          const v16h& a0, const v16h& a1,
                                          int lane, int t) {
  const float* row = ldsKf[t * 16 + (lane & 15)];
  v16h b0, b1;
#pragma unroll
  for (int e = 0; e < 16; ++e) {
    int d = ab_idx(lane, e);
    b0[e] = (_Float16)row[d];
    b1[e] = (_Float16)row[32 + d];
  }
  v8f c = {};
  c = __builtin_amdgcn_wmma_f32_16x16x32_f16(false, a0, false, b0, (short)0, c, false, false);
  c = __builtin_amdgcn_wmma_f32_16x16x32_f16(false, a1, false, b1, (short)0, c, false, false);
  return c;
}

// Issue the async K-tile copy: 32 rows x 64 dwords, 2 x b128 per thread.
__device__ __forceinline__ void stage_k_async(const float* __restrict__ Kbh, int kb,
                                              float (*buf)[LKF], int tid) {
  int kr = tid >> 3;              // 0..31
  int c0 = (tid & 7) * 8;         // dword column, step 8
  const float* g = Kbh + (size_t)(kb * KSTEP + kr) * Ddim + c0;
  unsigned l0 = (unsigned)(uintptr_t)&buf[kr][c0];
  async_copy_b128(l0, g);
  async_copy_b128(l0 + 16, g + 4);
}

// ---------------- Pass 1: per-row softmax stats (m, l) ----------------
__global__ __launch_bounds__(256)
void attn_stats_kernel(const float* __restrict__ Q, const float* __restrict__ K,
                       float* __restrict__ wsm, float* __restrict__ wsl) {
  __shared__ __align__(16) float ldsKf[2][KSTEP][LKF];   // ping-pong DMA buffers
  const int bh   = blockIdx.x / (Sdim / QBLK);
  const int qblk = blockIdx.x % (Sdim / QBLK);
  const int lane = threadIdx.x & 31;
  const int half = lane >> 4;
  const int ln   = lane & 15;
  const int qbase = qblk * QBLK + (threadIdx.x >> 5) * QT;

  const float* Qbh = Q + (size_t)bh * Sdim * Ddim;
  const float* Kbh = K + (size_t)bh * Sdim * Ddim;

  v16h a0, a1;
  load_q_frags(Qbh, qbase, lane, a0, a1);

  float m[8], l[8];
#pragma unroll
  for (int r = 0; r < 8; ++r) { m[r] = -3.0e38f; l[r] = 0.0f; }

  stage_k_async(Kbh, 0, ldsKf[0], threadIdx.x);          // prologue DMA

  for (int kb = 0; kb < NKB; ++kb) {
    const int cur = kb & 1;
    if (kb + 1 < NKB) {                                  // prefetch next tile
      stage_k_async(Kbh, kb + 1, ldsKf[cur ^ 1], threadIdx.x);
      wait_async2();                                     // tile kb landed (in-order)
    } else {
      wait_async0();
    }
    __syncthreads();

#pragma unroll
    for (int t = 0; t < 2; ++t) {
      v8f c = score_tile(ldsKf[cur], a0, a1, lane, t);
#pragma unroll
      for (int r = 0; r < 8; ++r) {       // per-lane online update, no shuffles
        float mo = m[r];
        float mn = fmaxf(mo, c[r]);
        l[r] = l[r] * __expf(mo - mn) + __expf(c[r] - mn);
        m[r] = mn;
      }
    }
    __syncthreads();                                     // protect buffer reuse
  }

  // single cross-lane merge of (m,l) pairs within each 16-lane half
#pragma unroll
  for (int r = 0; r < 8; ++r) {
#pragma unroll
    for (int mb = 1; mb < 16; mb <<= 1) {
      float mo = __shfl_xor(m[r], mb, 32);
      float lo = __shfl_xor(l[r], mb, 32);
      float mn = fmaxf(m[r], mo);
      l[r] = l[r] * __expf(m[r] - mn) + lo * __expf(mo - mn);
      m[r] = mn;
    }
  }

  if (ln == 0) {  // lanes 0 and 16 each own 8 rows
#pragma unroll
    for (int r = 0; r < 8; ++r) {
      int row = qbase + r + 8 * half;
      wsm[bh * Sdim + row] = m[r];
      wsl[bh * Sdim + row] = l[r];
    }
  }
}

// ---------- Pass 2: recompute scores, write attn once, O = P @ V ----------
__global__ __launch_bounds__(256)
void attn_out_kernel(const float* __restrict__ Q, const float* __restrict__ K,
                     const float* __restrict__ V,
                     const float* __restrict__ wsm, const float* __restrict__ wsl,
                     float* __restrict__ Out, float* __restrict__ Attn) {
  __shared__ __align__(16) float ldsKf[2][KSTEP][LKF];   // ping-pong DMA buffers
  __shared__ _Float16 ldsVt[2][Ddim][LVPAD];             // V transposed: [d][k-row]
  __shared__ _Float16 ldsP [WAVES][16][LPPAD];

  const int bh   = blockIdx.x / (Sdim / QBLK);
  const int qblk = blockIdx.x % (Sdim / QBLK);
  const int wave = threadIdx.x >> 5;
  const int lane = threadIdx.x & 31;
  const int half = lane >> 4;
  const int ln   = lane & 15;
  const int qbase = qblk * QBLK + wave * QT;

  const float* Qbh = Q + (size_t)bh * Sdim * Ddim;
  const float* Kbh = K + (size_t)bh * Sdim * Ddim;
  const float* Vbh = V + (size_t)bh * Sdim * Ddim;

  v16h a0, a1;
  load_q_frags(Qbh, qbase, lane, a0, a1);

  float m[8], li[8];
#pragma unroll
  for (int r = 0; r < 8; ++r) {
    int row = qbase + r + 8 * half;
    m[r]  = wsm[bh * Sdim + row];
    li[r] = 1.0f / wsl[bh * Sdim + row];
  }

  v8f o[4];
#pragma unroll
  for (int nt = 0; nt < 4; ++nt) o[nt] = (v8f){};

  // V staging geometry (per thread)
  const int vkr = threadIdx.x >> 3;
  const int vd0 = (threadIdx.x & 7) * 8;

  // prologue: DMA K0, prefetch V0 into registers
  stage_k_async(Kbh, 0, ldsKf[0], threadIdx.x);
  float vreg[8];
#pragma unroll
  for (int j = 0; j < 8; ++j) vreg[j] = Vbh[(size_t)vkr * Ddim + vd0 + j];

  for (int kb = 0; kb < NKB; ++kb) {
    const int cur = kb & 1;
    if (kb + 1 < NKB)                                    // async K prefetch
      stage_k_async(Kbh, kb + 1, ldsKf[cur ^ 1], threadIdx.x);

    // transpose-store prefetched V tile (f16) for this iteration
#pragma unroll
    for (int j = 0; j < 8; ++j) ldsVt[cur][vd0 + j][vkr] = (_Float16)vreg[j];

    if (kb + 1 < NKB) {                                  // prefetch next V tile into regs
      const float* sv = Vbh + (size_t)((kb + 1) * KSTEP + vkr) * Ddim + vd0;
#pragma unroll
      for (int j = 0; j < 8; ++j) vreg[j] = sv[j];
      wait_async2();                                     // K tile kb landed (in-order)
    } else {
      wait_async0();
    }
    __syncthreads();

#pragma unroll
    for (int t = 0; t < 2; ++t) {
      v8f c = score_tile(ldsKf[cur], a0, a1, lane, t);
#pragma unroll
      for (int r = 0; r < 8; ++r) c[r] = __expf(c[r] - m[r]) * li[r];
      // single write of normalized attn + stage P (f16) for the PV WMMA
#pragma unroll
      for (int r = 0; r < 8; ++r) {
        int qr = r + 8 * half;
        Attn[((size_t)bh * Sdim + qbase + qr) * Sdim + kb * KSTEP + t * 16 + ln] = c[r];
        ldsP[wave][qr][t * 16 + ln] = (_Float16)c[r];
      }
    }

    // C-layout -> A-layout through per-wave LDS (DS ops in-order within wave)
    v16h pa;
#pragma unroll
    for (int e = 0; e < 16; ++e) pa[e] = ldsP[wave][ln][ab_idx(lane, e)];

#pragma unroll
    for (int nt = 0; nt < 4; ++nt) {
      v16h b;   // B(k-row, n) = V[k-row][n] = ldsVt[n][k-row]; contiguous in e
#pragma unroll
      for (int e = 0; e < 16; ++e) b[e] = ldsVt[cur][nt * 16 + ln][ab_idx(lane, e)];
      o[nt] = __builtin_amdgcn_wmma_f32_16x16x32_f16(false, pa, false, b, (short)0,
                                                     o[nt], false, false);
    }
    __syncthreads();                                     // protect buffer reuse
  }

#pragma unroll
  for (int nt = 0; nt < 4; ++nt)
#pragma unroll
    for (int r = 0; r < 8; ++r)
      Out[((size_t)bh * Sdim + qbase + r + 8 * half) * Ddim + nt * 16 + ln] = o[nt][r];
}

extern "C" void kernel_launch(void* const* d_in, const int* in_sizes, int n_in,
                              void* d_out, int out_size, void* d_ws, size_t ws_size,
                              hipStream_t stream) {
  (void)in_sizes; (void)n_in; (void)out_size; (void)ws_size;
  const float* Q = (const float*)d_in[0];
  const float* K = (const float*)d_in[1];
  const float* V = (const float*)d_in[2];
  float* Out  = (float*)d_out;                                  // [B,H,S,D]
  float* Attn = Out + (size_t)Bdim * Hdim * Sdim * Ddim;        // [B,H,S,S]
  float* wsm  = (float*)d_ws;                                   // [BH,S] row max
  float* wsl  = wsm + (size_t)BHdim * Sdim;                     // [BH,S] row sumexp

  dim3 grid(BHdim * (Sdim / QBLK));   // 64 * 8 = 512 workgroups
  dim3 blk(256);                      // 8 wave32 waves
  attn_stats_kernel<<<grid, blk, 0, stream>>>(Q, K, wsm, wsl);
  attn_out_kernel<<<grid, blk, 0, stream>>>(Q, K, V, wsm, wsl, Out, Attn);
}
